// DecoderRNNWithAttention_71897752535758
// MI455X (gfx1250) — matrix operand order repaired and checked by reading, not compiled, not run on hardware
//
#include <hip/hip_runtime.h>
#include <hip/hip_bf16.h>

typedef __attribute__((ext_vector_type(16))) __bf16 v16bf;
typedef __attribute__((ext_vector_type(8)))  float  v8f;

union FragBF {
    v16bf v;
    __bf16 h[16];
    unsigned int u[8];
};

__device__ __forceinline__ float sigmoidf_(float x) { return 1.0f / (1.0f + __expf(-x)); }

// ---------------- conversions / init ----------------
__global__ void k_conv_bf16(const float* __restrict__ src, __bf16* __restrict__ dst, int n) {
    int i = blockIdx.x * blockDim.x + threadIdx.x;
    if (i < n) dst[i] = (__bf16)src[i];
}

// src is rows x cols (row-major), dst becomes cols x rows (row-major)
__global__ void k_conv_bf16_T(const float* __restrict__ src, __bf16* __restrict__ dst,
                              int rows, int cols) {
    int i = blockIdx.x * blockDim.x + threadIdx.x;
    if (i >= rows * cols) return;
    int r = i / cols, c = i % cols;
    dst[(size_t)c * rows + r] = (__bf16)src[i];
}

__global__ void k_zero(float* __restrict__ p, int n) {
    int i = blockIdx.x * blockDim.x + threadIdx.x;
    if (i < n) p[i] = 0.0f;
}

__global__ void k_zero_bf16(__bf16* __restrict__ p, int n) {
    int i = blockIdx.x * blockDim.x + threadIdx.x;
    if (i < n) p[i] = (__bf16)0.0f;
}

// ---------------- WMMA GEMM (NT-strip): C[M,N] = A[M,K](bf16) x W[N,K]^T(bf16) + bias ----------------
// One wave computes a 16 x (16*NT) strip: A fragment loaded once per k-step, reused for NT WMMAs.
// EPI 0: C = val+bias ; EPI 1: C += val+bias ; EPI 2: C = (t<lengths[row]) ? val+bias : 0
template<int EPI, int NT>
__global__ __launch_bounds__(256)
void k_gemm_nt(const __bf16* __restrict__ A, const __bf16* __restrict__ W,
               const float* __restrict__ bias, float* __restrict__ C,
               int M, int N, int K, int ldc, const int* __restrict__ lengths, int t)
{
    const int lane   = threadIdx.x & 31;
    const int wave   = threadIdx.x >> 5;
    const int groups = (N >> 4) / NT;                 // N-tile groups per M row
    const int tile   = blockIdx.x * (blockDim.x >> 5) + wave;
    if (tile >= (M >> 4) * groups) return;            // wave-uniform: EXEC stays all-1s
    const int tm = (tile / groups) << 4;
    const int tn = (tile % groups) * (NT << 4);

    const int mn_lane = lane & 15;
    const int half    = lane >> 4;
    const __bf16* arow = A + (size_t)(tm + mn_lane) * K;
    const __bf16* wrow[NT];
    #pragma unroll
    for (int j = 0; j < NT; ++j)
        wrow[j] = W + (size_t)(tn + (j << 4) + mn_lane) * K;

    v8f acc[NT] = {};
    for (int k0 = 0; k0 < K; k0 += 32) {
        FragBF a;
        #pragma unroll
        for (int v = 0; v < 8; ++v) {
            // A 16x32 bf16 fragment: lane half selects K block, VGPR v -> K pair
            const int ka = k0 + ((v & 3) << 1) + ((v >> 2) << 4) + (half << 3);
            a.u[v] = *reinterpret_cast<const unsigned int*>(arow + ka);
        }
        #pragma unroll
        for (int j = 0; j < NT; ++j) {
            FragBF b;
            #pragma unroll
            for (int v = 0; v < 8; ++v) {
                // B 32x16 bf16 fragment: lane half selects K block, VGPR v -> K pair (same as A)
                const int kb = k0 + ((v & 3) << 1) + ((v >> 2) << 4) + (half << 3);
                b.u[v] = *reinterpret_cast<const unsigned int*>(wrow[j] + kb);
            }
            acc[j] = __builtin_amdgcn_wmma_f32_16x16x32_bf16(false, a.v, false, b.v,
                                                             (short)0, acc[j], false, false);
        }
    }

    #pragma unroll
    for (int j = 0; j < NT; ++j) {
        const int n  = tn + (j << 4) + mn_lane;
        const float bv = bias ? bias[n] : 0.0f;
        #pragma unroll
        for (int v = 0; v < 8; ++v) {
            const int row = tm + v + (half << 3);     // C/D layout: VGPR v -> M row
            float val = acc[j][v] + bv;
            if (EPI == 0) {
                C[(size_t)row * ldc + n] = val;
            } else if (EPI == 1) {
                C[(size_t)row * ldc + n] += val;
            } else {
                const bool m = t < lengths[row];
                C[(size_t)row * ldc + n] = m ? val : 0.0f;
            }
        }
    }
}

// ---------------- attention energy: e[b,l] = relu(fp[b,l,:] + S[b,:]) . w_att + b_att ----------------
__global__ __launch_bounds__(256)
void k_att_energy(const float* __restrict__ feat_proj, const float* __restrict__ S,
                  const float* __restrict__ w_att, const float* __restrict__ b_att,
                  float* __restrict__ e, int Bn, int Ln, int An)
{
    const int gw   = (blockIdx.x * blockDim.x + threadIdx.x) >> 5;
    const int lane = threadIdx.x & 31;
    if (gw >= Bn * Ln) return;
    const int b = gw / Ln;
    const float* fp = feat_proj + (size_t)gw * An;
    const float* sr = S + (size_t)b * An;
    float sum = 0.0f;
    for (int a = lane; a < An; a += 32) {
        float v = fp[a] + sr[a];
        v = v > 0.0f ? v : 0.0f;
        sum += v * w_att[a];
    }
    #pragma unroll
    for (int o = 16; o; o >>= 1) sum += __shfl_xor(sum, o);
    if (lane == 0) e[gw] = sum + b_att[0];
}

// ---------------- softmax over L per batch row (in place) ----------------
__global__ __launch_bounds__(256)
void k_softmax(float* __restrict__ e, int Bn, int Ln)
{
    const int gw   = (blockIdx.x * blockDim.x + threadIdx.x) >> 5;
    const int lane = threadIdx.x & 31;
    if (gw >= Bn) return;
    float* row = e + (size_t)gw * Ln;
    float v0 = (lane      < Ln) ? row[lane]      : -3.0e38f;
    float v1 = (lane + 32 < Ln) ? row[lane + 32] : -3.0e38f;
    float mx = fmaxf(v0, v1);
    #pragma unroll
    for (int o = 16; o; o >>= 1) mx = fmaxf(mx, __shfl_xor(mx, o));
    float e0 = (lane      < Ln) ? __expf(v0 - mx) : 0.0f;
    float e1 = (lane + 32 < Ln) ? __expf(v1 - mx) : 0.0f;
    float s = e0 + e1;
    #pragma unroll
    for (int o = 16; o; o >>= 1) s += __shfl_xor(s, o);
    const float inv = 1.0f / s;
    if (lane      < Ln) row[lane]      = e0 * inv;
    if (lane + 32 < Ln) row[lane + 32] = e1 * inv;
}

// ---------------- context[b,d] = sum_l alpha[b,l] * features[b,l,d] ----------------
__global__ __launch_bounds__(256)
void k_context(const float* __restrict__ alpha, const float* __restrict__ features,
               float* __restrict__ ctx, int Ln, int Dn)
{
    __shared__ float al[64];
    const int b = blockIdx.y;
    const int d = blockIdx.x * blockDim.x + threadIdx.x;
    if ((int)threadIdx.x < Ln) al[threadIdx.x] = alpha[(size_t)b * Ln + threadIdx.x];
    __syncthreads();
    const float* f = features + (size_t)b * Ln * Dn + d;
    float acc = 0.0f;
    for (int l = 0; l < Ln; ++l) acc += al[l] * f[(size_t)l * Dn];
    ctx[(size_t)b * Dn + d] = acc;
}

// ---------------- x_bf[b,:] = bf16(concat(embed_table[captions[b,t]], context[b,:])) ----------------
__global__ __launch_bounds__(256)
void k_build_x(const float* __restrict__ embed, const int* __restrict__ captions,
               const float* __restrict__ ctx, __bf16* __restrict__ x_bf,
               int t, int Tn, int En, int Dn)
{
    const int b = blockIdx.y;
    const int i = blockIdx.x * blockDim.x + threadIdx.x;
    float v;
    if (i < En) {
        const int tok = captions[b * Tn + t];
        v = embed[(size_t)tok * En + i];
    } else {
        v = ctx[(size_t)b * Dn + (i - En)];
    }
    x_bf[(size_t)b * (En + Dn) + i] = (__bf16)v;
}

// ---------------- LSTM cell + masked state update (writes bf16 mirrors for the GEMMs) ----------------
__global__ __launch_bounds__(256)
void k_lstm_cell(const float* __restrict__ gates, float* __restrict__ h, float* __restrict__ c,
                 __bf16* __restrict__ h_bf, __bf16* __restrict__ hnew_bf,
                 const int* __restrict__ lengths, int t, int Hn)
{
    const int idx = blockIdx.x * blockDim.x + threadIdx.x;
    const int b = idx / Hn, j = idx % Hn;
    const float* g = gates + (size_t)b * 4 * Hn;
    const float ig = sigmoidf_(g[j]);
    const float fg = sigmoidf_(g[Hn + j]);
    const float gg = tanhf(g[2 * Hn + j]);
    const float og = sigmoidf_(g[3 * Hn + j]);
    const float cn = fg * c[idx] + ig * gg;
    const float hn = og * tanhf(cn);
    hnew_bf[idx] = (__bf16)hn;                       // logits always use h_new
    const bool m = t < lengths[b];
    const float hkeep = m ? hn : h[idx];
    if (m) { h[idx] = hn; c[idx] = cn; }
    h_bf[idx] = (__bf16)hkeep;                       // next step's GEMM input
}

// ---------------- launcher ----------------
extern "C" void kernel_launch(void* const* d_in, const int* in_sizes, int n_in,
                              void* d_out, int out_size, void* d_ws, size_t ws_size,
                              hipStream_t stream)
{
    constexpr int Bn = 128, Tn = 32, Vn = 10000, En = 256, Hn = 512, An = 512, Dn = 2048, Ln = 49;
    const float* features = (const float*)d_in[0];
    const int*   captions = (const int*)  d_in[1];
    const int*   lengths  = (const int*)  d_in[2];
    const float* embed    = (const float*)d_in[3];
    const float* W_enc    = (const float*)d_in[4];
    const float* b_enc    = (const float*)d_in[5];
    const float* W_dec    = (const float*)d_in[6];
    const float* b_dec    = (const float*)d_in[7];
    const float* w_att    = (const float*)d_in[8];
    const float* b_att    = (const float*)d_in[9];
    const float* W_ih     = (const float*)d_in[10];
    const float* W_hh     = (const float*)d_in[11];
    const float* b_ih     = (const float*)d_in[12];
    const float* b_hh     = (const float*)d_in[13];
    const float* W_out    = (const float*)d_in[14];
    const float* b_out    = (const float*)d_in[15];
    float* out = (float*)d_out;

    // workspace carve-up (~66 MB)
    char* p = (char*)d_ws;
    auto alloc = [&](size_t bytes) { char* r = p; p += (bytes + 255) & ~size_t(255); return r; };
    __bf16* W_encT  = (__bf16*)alloc((size_t)An * Dn * 2);            // (A,D)
    __bf16* W_decT  = (__bf16*)alloc((size_t)An * Hn * 2);            // (A,H)
    __bf16* W_ih_b  = (__bf16*)alloc((size_t)4 * Hn * (En + Dn) * 2);
    __bf16* W_hh_b  = (__bf16*)alloc((size_t)4 * Hn * Hn * 2);
    __bf16* W_out_b = (__bf16*)alloc((size_t)Vn * Hn * 2);
    __bf16* feat_bf = (__bf16*)alloc((size_t)Bn * Ln * Dn * 2);       // bf16 features for feat_proj GEMM
    __bf16* x_bf    = (__bf16*)alloc((size_t)Bn * (En + Dn) * 2);
    __bf16* h_bf    = (__bf16*)alloc((size_t)Bn * Hn * 2);
    __bf16* hnew_bf = (__bf16*)alloc((size_t)Bn * Hn * 2);
    float* feat_proj = (float*)alloc((size_t)Bn * Ln * An * 4);
    float* S      = (float*)alloc((size_t)Bn * An * 4);
    float* e      = (float*)alloc((size_t)Bn * Ln * 4);
    float* ctx    = (float*)alloc((size_t)Bn * Dn * 4);
    float* gates  = (float*)alloc((size_t)Bn * 4 * Hn * 4);
    float* h      = (float*)alloc((size_t)Bn * Hn * 4);
    float* c      = (float*)alloc((size_t)Bn * Hn * 4);

    const int TPB = 256;
    auto gb = [](int n) { return (n + 255) / 256; };
    auto gemm_blocks = [](int M, int N, int NT) {
        int tiles = (M / 16) * ((N / 16) / NT);
        return (tiles + 7) / 8;
    };

    // one-time (per launch) staging to bf16; weights stay L2-resident (192 MB) for all 32 steps
    k_conv_bf16_T<<<gb(Dn * An), TPB, 0, stream>>>(W_enc, W_encT, Dn, An);
    k_conv_bf16_T<<<gb(Hn * An), TPB, 0, stream>>>(W_dec, W_decT, Hn, An);
    k_conv_bf16<<<gb(4 * Hn * (En + Dn)), TPB, 0, stream>>>(W_ih, W_ih_b, 4 * Hn * (En + Dn));
    k_conv_bf16<<<gb(4 * Hn * Hn), TPB, 0, stream>>>(W_hh, W_hh_b, 4 * Hn * Hn);
    k_conv_bf16<<<gb(Vn * Hn), TPB, 0, stream>>>(W_out, W_out_b, Vn * Hn);
    k_conv_bf16<<<gb(Bn * Ln * Dn), TPB, 0, stream>>>(features, feat_bf, Bn * Ln * Dn);
    k_zero<<<gb(Bn * Hn), TPB, 0, stream>>>(h, Bn * Hn);
    k_zero<<<gb(Bn * Hn), TPB, 0, stream>>>(c, Bn * Hn);
    k_zero_bf16<<<gb(Bn * Hn), TPB, 0, stream>>>(h_bf, Bn * Hn);

    // feat_proj = features @ W_enc + b_enc   (6272 x 512, K=2048)
    k_gemm_nt<0, 4><<<gemm_blocks(Bn * Ln, An, 4), TPB, 0, stream>>>(
        feat_bf, W_encT, b_enc, feat_proj, Bn * Ln, An, Dn, An, nullptr, 0);

    for (int t = 0; t < Tn; ++t) {
        // S = h @ W_dec + b_dec  (128 x 512, K=512)
        k_gemm_nt<0, 4><<<gemm_blocks(Bn, An, 4), TPB, 0, stream>>>(
            h_bf, W_decT, b_dec, S, Bn, An, Hn, An, nullptr, 0);
        // e = relu(feat_proj + S) . w_att + b_att
        k_att_energy<<<gb(Bn * Ln * 32), TPB, 0, stream>>>(feat_proj, S, w_att, b_att, e, Bn, Ln, An);
        // alpha = softmax(e) over L
        k_softmax<<<gb(Bn * 32), TPB, 0, stream>>>(e, Bn, Ln);
        // context = alpha @ features
        k_context<<<dim3(Dn / TPB, Bn), TPB, 0, stream>>>(e, features, ctx, Ln, Dn);
        // x = concat(emb_t, context) -> bf16
        k_build_x<<<dim3((En + Dn) / TPB, Bn), TPB, 0, stream>>>(embed, captions, ctx, x_bf, t, Tn, En, Dn);
        // gates = x @ W_ih^T + b_ih  (128 x 2048, K=2304)
        k_gemm_nt<0, 4><<<gemm_blocks(Bn, 4 * Hn, 4), TPB, 0, stream>>>(
            x_bf, W_ih_b, b_ih, gates, Bn, 4 * Hn, En + Dn, 4 * Hn, nullptr, 0);
        // gates += h @ W_hh^T + b_hh  (K=512)
        k_gemm_nt<1, 4><<<gemm_blocks(Bn, 4 * Hn, 4), TPB, 0, stream>>>(
            h_bf, W_hh_b, b_hh, gates, Bn, 4 * Hn, Hn, 4 * Hn, nullptr, 0);
        // LSTM cell + masked h/c update (+ bf16 mirrors)
        k_lstm_cell<<<gb(Bn * Hn), TPB, 0, stream>>>(gates, h, c, h_bf, hnew_bf, lengths, t, Hn);
        // logits -> masked write to out[:, t, :]  (128 x 10000, K=512; 625 N-tiles = 125 strips of 5)
        k_gemm_nt<2, 5><<<gemm_blocks(Bn, Vn, 5), TPB, 0, stream>>>(
            hnew_bf, W_out_b, b_out, out + (size_t)t * Vn, Bn, Vn, Hn, Tn * Vn, lengths, t);
    }
}
